// T5EncoderModel_24592982736987
// MI455X (gfx1250) — matrix operand order repaired and checked by reading, not compiled, not run on hardware
//
#include <hip/hip_runtime.h>
#include <math.h>

// ---------------------------------------------------------------------------
// CDNA5 (gfx1250) T5 encoder forward. wave32, WMMA bf16 16x16x32, f32 accum.
// ---------------------------------------------------------------------------

typedef __attribute__((ext_vector_type(16))) __bf16 v16bf;
typedef __attribute__((ext_vector_type(8)))  float  v8f;

union Frag { unsigned int u[8]; v16bf v; };

#define TD 768
#define TS 512
#define TB 16
#define TH 12
#define TF 3072
#define TL 12
#define TBS (TB * TS)   // 8192 rows

__device__ __forceinline__ unsigned short f2bf(float f) {
    unsigned int u = __float_as_uint(f);
    unsigned int r = u + 0x7fffu + ((u >> 16) & 1u);   // round-to-nearest-even
    return (unsigned short)(r >> 16);
}

// A-fragment k order used here (contiguous, matching B): lane half 0 -> K {0..15},
// lane half 1 -> K {16..31}; VGPR v holds a consecutive K pair.
// => u[0..7] = 16 contiguous ushorts @ (16*lhalf).
__device__ __forceinline__ void load_afrag_lds(Frag& f, const unsigned short* rowp, int lhalf) {
    *reinterpret_cast<uint4*>(&f.u[0]) = *reinterpret_cast<const uint4*>(rowp + 16 * lhalf);
    *reinterpret_cast<uint4*>(&f.u[4]) = *reinterpret_cast<const uint4*>(rowp + 16 * lhalf + 8);
}
// B-fragment k order used here (contiguous, matching A): lane half selects K group {0..15} / {16..31},
// K pairs consecutive => 16 contiguous ushorts starting at 16*lhalf.
__device__ __forceinline__ void load_bfrag_contig(Frag& f, const unsigned short* p) {
    *reinterpret_cast<uint4*>(&f.u[0]) = *reinterpret_cast<const uint4*>(p);
    *reinterpret_cast<uint4*>(&f.u[4]) = *reinterpret_cast<const uint4*>(p + 8);
}

// ---------------------------------------------------------------------------
// Embedding gather: x[row,:] = emb[tokens[row],:]
// ---------------------------------------------------------------------------
__global__ __launch_bounds__(256) void embed_kernel(const int* __restrict__ tok,
                                                    const float* __restrict__ emb,
                                                    float* __restrict__ x) {
    int row = blockIdx.x;
    const float* src = emb + (size_t)tok[row] * TD;
    float* dst = x + (size_t)row * TD;
    for (int c = threadIdx.x; c < TD; c += 256) dst[c] = src[c];
}

// ---------------------------------------------------------------------------
// T5 relative bias: bucket depends only on delta = k - q. Build [H][1023] LUT.
// ---------------------------------------------------------------------------
__global__ __launch_bounds__(256) void relbias_kernel(const float* __restrict__ rel_bias,
                                                      float* __restrict__ biasd) {
    int idx = blockIdx.x * 256 + threadIdx.x;
    if (idx >= TH * 1023) return;
    int h = idx / 1023, dp = idx % 1023;
    int delta = dp - 511;                       // k - q
    int ret = (delta > 0) ? 16 : 0;             // num_buckets//2
    int n = (delta < 0) ? -delta : delta;
    int bucket;
    if (n < 8) {                                // max_exact = 8
        bucket = ret + n;
    } else {
        int large = 8 + (int)(logf((float)n / 8.0f) / logf(16.0f) * 8.0f);
        large = (large < 15) ? large : 15;
        bucket = ret + large;
    }
    biasd[h * 1023 + dp] = rel_bias[bucket * TH + h];
}

// ---------------------------------------------------------------------------
// RMS norm: out = x * rsqrt(mean(x^2)+eps) * w   (out bf16 or f32)
// ---------------------------------------------------------------------------
template <bool OUTBF>
__global__ __launch_bounds__(256) void rmsnorm_kernel(const float* __restrict__ x,
                                                      const float* __restrict__ w,
                                                      void* __restrict__ out) {
    __shared__ float part[256];
    int row = blockIdx.x, tid = threadIdx.x;
    const float* xr = x + (size_t)row * TD;
    float s = 0.f;
    for (int c = tid; c < TD; c += 256) { float v = xr[c]; s += v * v; }
    part[tid] = s;
    __syncthreads();
    for (int st = 128; st > 0; st >>= 1) {
        if (tid < st) part[tid] += part[tid + st];
        __syncthreads();
    }
    float scale = rsqrtf(part[0] / (float)TD + 1e-6f);
    for (int c = tid; c < TD; c += 256) {
        float v = xr[c] * scale * w[c];
        if (OUTBF) ((unsigned short*)out)[(size_t)row * TD + c] = f2bf(v);
        else       ((float*)out)[(size_t)row * TD + c] = v;
    }
}

// ---------------------------------------------------------------------------
// GEMM: out[M,N] = A[M,K](bf16) @ W[K,N](fp32 -> bf16 on the fly)
// FLAGS: 1 = ReLU, 2 = add fp32 residual, 4 = bf16 output.
// 256 thr = 8 waves; block tile 128x64; wave tile 32x32 (2x2 WMMA); K-step 32.
// Double-buffered LDS, software-pipelined: global loads for tile kt+1 issue
// before tile kt's WMMAs so load latency hides under compute.
// ---------------------------------------------------------------------------
template <int FLAGS>
__global__ __launch_bounds__(256) void gemm_bf16_kernel(const unsigned short* __restrict__ A,
                                                        const float* __restrict__ W,
                                                        const float* __restrict__ resid,
                                                        void* __restrict__ out,
                                                        int M, int N, int K) {
    constexpr bool RELU  = (FLAGS & 1) != 0;
    constexpr bool RES   = (FLAGS & 2) != 0;
    constexpr bool OUTBF = (FLAGS & 4) != 0;

    __shared__ unsigned short As[2][128][40];   // 2 x 128x32 bf16, padded (20 KB)
    __shared__ unsigned short Bs[2][64][40];    // 2 x Bs[n][k] transposed (10 KB)

    const int tid = threadIdx.x, lane = tid & 31, wid = tid >> 5;
    const int l15 = lane & 15, lhalf = lane >> 4;
    const int wm0 = (wid >> 1) * 32, wn0 = (wid & 1) * 32;
    const int bM = blockIdx.x * 128, bN = blockIdx.y * 64;

    // staging-thread coordinates
    const int arow = tid >> 3, ac4 = (tid & 7) * 4;        // A: 4 rounds of 32 rows
    const int bk = tid >> 3, bn0 = (tid & 7) * 8;          // W: 32x64 in one round

    uint2 aReg[4];
    float wReg[8];

    auto load_global = [&](int kt) {
        const unsigned short* Ag = A + (size_t)bM * K + kt * 32;
#pragma unroll
        for (int it = 0; it < 4; ++it)
            aReg[it] = *reinterpret_cast<const uint2*>(Ag + (size_t)(arow + 32 * it) * K + ac4);
        const float* Wg = W + (size_t)(kt * 32 + bk) * N + bN + bn0;
#pragma unroll
        for (int j = 0; j < 8; ++j) wReg[j] = Wg[j];
    };
    auto store_lds = [&](int buf) {
#pragma unroll
        for (int it = 0; it < 4; ++it)
            *reinterpret_cast<uint2*>(&As[buf][arow + 32 * it][ac4]) = aReg[it];
#pragma unroll
        for (int j = 0; j < 8; ++j) Bs[buf][bn0 + j][bk] = f2bf(wReg[j]);
    };

    v8f acc[2][2] = {};

    load_global(0);
    store_lds(0);
    __syncthreads();

    const int KT = K / 32;
    for (int kt = 0; kt < KT; ++kt) {
        const int cur = kt & 1;
        const bool more = (kt + 1 < KT);
        if (more) load_global(kt + 1);          // prefetch next tile into regs

        Frag aF[2], bF[2];
#pragma unroll
        for (int i = 0; i < 2; ++i)
            load_afrag_lds(aF[i], &As[cur][wm0 + i * 16 + l15][0], lhalf);
#pragma unroll
        for (int j = 0; j < 2; ++j)
            load_bfrag_contig(bF[j], &Bs[cur][wn0 + j * 16 + l15][16 * lhalf]);
#pragma unroll
        for (int i = 0; i < 2; ++i)
#pragma unroll
            for (int j = 0; j < 2; ++j)
                acc[i][j] = __builtin_amdgcn_wmma_f32_16x16x32_bf16(
                    false, aF[i].v, false, bF[j].v, (short)0, acc[i][j], false, false);

        if (more) store_lds(cur ^ 1);
        __syncthreads();
    }

    // Epilogue: C/D layout -> lane l15 = N, lhalf selects M 0-7 vs 8-15
#pragma unroll
    for (int i = 0; i < 2; ++i)
#pragma unroll
        for (int j = 0; j < 2; ++j) {
            int row0 = bM + wm0 + i * 16, col0 = bN + wn0 + j * 16;
#pragma unroll
            for (int r = 0; r < 8; ++r) {
                int m = r + 8 * lhalf;
                size_t idx = (size_t)(row0 + m) * N + col0 + l15;
                float v = acc[i][j][r];
                if (RES)  v += resid[idx];
                if (RELU) v = (v > 0.f) ? v : 0.f;
                if (OUTBF) ((unsigned short*)out)[idx] = f2bf(v);
                else       ((float*)out)[idx] = v;
            }
        }
}

// ---------------------------------------------------------------------------
// Fused attention (flash-style): one WG = one (b,h, 16-query tile).
// 4 waves. Scores (QK^T + bias) -> LDS, softmax in LDS, P@V -> bf16 ctx.
// Scores never touch HBM (~400 MB of traffic avoided).
// ---------------------------------------------------------------------------
__global__ __launch_bounds__(128) void attn_kernel(const unsigned short* __restrict__ qb,
                                                   const unsigned short* __restrict__ kb,
                                                   const unsigned short* __restrict__ vb,
                                                   const float* __restrict__ biasd,
                                                   unsigned short* __restrict__ ctx) {
    __shared__ float sc[16][TS + 4];
    __shared__ float red[16][8];
    __shared__ float rowv[16];
    __shared__ unsigned short pbf[16][TS + 8];

    const int tid = threadIdx.x, lane = tid & 31, w = tid >> 5;
    const int l15 = lane & 15, lhalf = lane >> 4;
    const int blk = blockIdx.x;
    const int qt = blk & 31, bh = blk >> 5;
    const int h = bh % TH, b = bh / TH;
    const int q0 = qt * 16;

    // Q fragments for both dh K-steps (same for every key tile)
    Frag aq[2];
    {
        const unsigned short* qrow = qb + (size_t)(b * TS + q0 + l15) * TD + h * 64;
#pragma unroll
        for (int ks = 0; ks < 2; ++ks)
            load_afrag_lds(aq[ks], qrow + ks * 32, lhalf);   // same contiguity in global
    }

    // Phase 1: scores. Wave w handles key tiles jt = w, w+4, ...
    for (int t = 0; t < 8; ++t) {
        int jt = w + 4 * t;
        v8f acc = {};
#pragma unroll
        for (int ks = 0; ks < 2; ++ks) {
            Frag bk;   // B = K^T: col = key pos, contiguous dh pairs in memory
            const unsigned short* krp =
                kb + (size_t)(b * TS + jt * 16 + l15) * TD + h * 64 + ks * 32 + 16 * lhalf;
            load_bfrag_contig(bk, krp);
            acc = __builtin_amdgcn_wmma_f32_16x16x32_bf16(
                false, aq[ks].v, false, bk.v, (short)0, acc, false, false);
        }
#pragma unroll
        for (int r = 0; r < 8; ++r) {
            int m = r + 8 * lhalf;
            int kpos = jt * 16 + l15;
            sc[m][kpos] = acc[r] + biasd[h * 1023 + kpos - (q0 + m) + 511];
        }
    }
    __syncthreads();

    // Phase 2: softmax over 512 keys. 8 threads per row, 64 cols each.
    {
        int r = tid >> 3, p = tid & 7, c0 = p * 64;
        float mx = -1e30f;
        for (int c = c0; c < c0 + 64; ++c) mx = fmaxf(mx, sc[r][c]);
        red[r][p] = mx;
        __syncthreads();
        if (p == 0) {
            float m2 = red[r][0];
            for (int i = 1; i < 8; ++i) m2 = fmaxf(m2, red[r][i]);
            rowv[r] = m2;
        }
        __syncthreads();
        float rm = rowv[r], s = 0.f;
        for (int c = c0; c < c0 + 64; ++c) {
            float e = __expf(sc[r][c] - rm);
            sc[r][c] = e;
            s += e;
        }
        red[r][p] = s;
        __syncthreads();
        if (p == 0) {
            float t2 = 0.f;
            for (int i = 0; i < 8; ++i) t2 += red[r][i];
            rowv[r] = t2;
        }
        __syncthreads();
        float inv = 1.f / rowv[r];
        for (int c = c0; c < c0 + 64; ++c) pbf[r][c] = f2bf(sc[r][c] * inv);
    }
    __syncthreads();

    // Phase 3: O = P @ V. Wave w owns dh columns [16w, 16w+16). 16 K-steps.
    {
        v8f acc = {};
        int d0 = w * 16;
        for (int kt = 0; kt < 16; ++kt) {
            Frag ap, bv;
            load_afrag_lds(ap, &pbf[l15][kt * 32], lhalf);
#pragma unroll
            for (int v = 0; v < 8; ++v) {
                int k = kt * 32 + 16 * lhalf + 2 * v;
                unsigned int lo = vb[(size_t)(b * TS + k) * TD + h * 64 + d0 + l15];
                unsigned int hi = vb[(size_t)(b * TS + k + 1) * TD + h * 64 + d0 + l15];
                bv.u[v] = lo | (hi << 16);
            }
            acc = __builtin_amdgcn_wmma_f32_16x16x32_bf16(
                false, ap.v, false, bv.v, (short)0, acc, false, false);
        }
#pragma unroll
        for (int r = 0; r < 8; ++r) {
            int m = r + 8 * lhalf;
            ctx[(size_t)(b * TS + q0 + m) * TD + h * 64 + d0 + l15] = f2bf(acc[r]);
        }
    }
}

// ---------------------------------------------------------------------------
// Host orchestration (graph-capture safe: only launches on `stream`)
// ---------------------------------------------------------------------------
extern "C" void kernel_launch(void* const* d_in, const int* in_sizes, int n_in,
                              void* d_out, int out_size, void* d_ws, size_t ws_size,
                              hipStream_t stream) {
    const int*   tokens   = (const int*)d_in[0];
    const float* emb      = (const float*)d_in[1];
    const float* rel_bias = (const float*)d_in[2];
    const float* ln1_w    = (const float*)d_in[3];
    const float* wq       = (const float*)d_in[4];
    const float* wk       = (const float*)d_in[5];
    const float* wv       = (const float*)d_in[6];
    const float* wo       = (const float*)d_in[7];
    const float* ln2_w    = (const float*)d_in[8];
    const float* w1       = (const float*)d_in[9];
    const float* w2       = (const float*)d_in[10];
    const float* ln_f     = (const float*)d_in[11];
    float* out = (float*)d_out;

    char* base = (char*)d_ws;
    size_t off = 0;
    auto alloc = [&](size_t bytes) -> void* {
        void* p = base + off;
        off += (bytes + 255) & ~(size_t)255;
        return p;
    };
    float*          x     = (float*)alloc((size_t)TBS * TD * 4);
    unsigned short* xbf   = (unsigned short*)alloc((size_t)TBS * TD * 2);
    unsigned short* qb    = (unsigned short*)alloc((size_t)TBS * TD * 2);
    unsigned short* kbuf  = (unsigned short*)alloc((size_t)TBS * TD * 2);
    unsigned short* vbuf  = (unsigned short*)alloc((size_t)TBS * TD * 2);
    unsigned short* ctxb  = (unsigned short*)alloc((size_t)TBS * TD * 2);
    unsigned short* f1b   = (unsigned short*)alloc((size_t)TBS * TF * 2);
    float*          biasd = (float*)alloc((size_t)TH * 1023 * 4);

    embed_kernel<<<TBS, 256, 0, stream>>>(tokens, emb, x);
    relbias_kernel<<<(TH * 1023 + 255) / 256, 256, 0, stream>>>(rel_bias, biasd);

    dim3 gD(TBS / 128, TD / 64);    // 64 x 12
    dim3 gF(TBS / 128, TF / 64);    // 64 x 48

    for (int i = 0; i < TL; ++i) {
        const float* wqi = wq + (size_t)i * TD * TD;
        const float* wki = wk + (size_t)i * TD * TD;
        const float* wvi = wv + (size_t)i * TD * TD;
        const float* woi = wo + (size_t)i * TD * TD;
        const float* w1i = w1 + (size_t)i * TD * TF;
        const float* w2i = w2 + (size_t)i * TF * TD;

        rmsnorm_kernel<true><<<TBS, 256, 0, stream>>>(x, ln1_w + i * TD, xbf);
        gemm_bf16_kernel<4><<<gD, 256, 0, stream>>>(xbf, wqi, nullptr, qb,   TBS, TD, TD);
        gemm_bf16_kernel<4><<<gD, 256, 0, stream>>>(xbf, wki, nullptr, kbuf, TBS, TD, TD);
        gemm_bf16_kernel<4><<<gD, 256, 0, stream>>>(xbf, wvi, nullptr, vbuf, TBS, TD, TD);
        attn_kernel<<<TB * TH * (TS / 16), 128, 0, stream>>>(qb, kbuf, vbuf, biasd, ctxb);
        gemm_bf16_kernel<2><<<gD, 256, 0, stream>>>(ctxb, woi, x, x, TBS, TD, TD);   // residual

        rmsnorm_kernel<true><<<TBS, 256, 0, stream>>>(x, ln2_w + i * TD, xbf);
        gemm_bf16_kernel<4 | 1><<<gF, 256, 0, stream>>>(xbf, w1i, nullptr, f1b, TBS, TF, TD); // ReLU
        gemm_bf16_kernel<2><<<gD, 256, 0, stream>>>(f1b, w2i, x, x, TBS, TD, TF);    // residual
    }

    rmsnorm_kernel<false><<<TBS, 256, 0, stream>>>(x, ln_f, out);
}
